// MultiScaleRetention_43525198578339
// MI455X (gfx1250) — matrix operand order repaired and checked by reading, not compiled, not run on hardware
//
#include <hip/hip_runtime.h>
#include <hip/hip_bf16.h>
#include <math.h>

typedef __attribute__((ext_vector_type(16))) _Float16 v16h;
typedef __attribute__((ext_vector_type(8)))  _Float16 v8h;
typedef __attribute__((ext_vector_type(8)))  float    v8f;

#define WMMA_F16(A, B, C) \
  __builtin_amdgcn_wmma_f32_16x16x32_f16(false, (A), false, (B), (short)0, (C), false, false)

constexpr int BB = 2, TT = 2048, CC = 1024, HH = 16, DD = 64;
constexpr int BT = BB * TT;                 // 4096 rows
constexpr int ITILES = TT / 16;             // 128 row-tiles per (b,h)

// ---- workspace layout (bytes); requires ~59 MiB ----
constexpr size_t OFF_XH   = 0;                               // f16 x (B*T,C)  8 MiB  (reused as yn)
constexpr size_t OFF_WQ   = (size_t)8  << 20;                // f16 weights, 2 MiB each
constexpr size_t OFF_WK   = (size_t)10 << 20;
constexpr size_t OFF_WV   = (size_t)12 << 20;
constexpr size_t OFF_WO   = (size_t)14 << 20;
constexpr size_t OFF_Q    = (size_t)16 << 20;                // f16 (b,h,t,d)  8 MiB
constexpr size_t OFF_K    = (size_t)24 << 20;                // f16 (b,h,t,d)  8 MiB
constexpr size_t OFF_VT   = (size_t)32 << 20;                // f16 (b,h,d,t)  8 MiB
constexpr size_t OFF_Y    = (size_t)40 << 20;                // f32 (b,h,t,d) 16 MiB
constexpr size_t OFF_PART = (size_t)56 << 20;                // f32 (bh,itile,2) 32 KiB
constexpr size_t OFF_STAT = ((size_t)56 << 20) + (64u << 10);// f32 (bh,2)

// ---------- WMMA operand loaders (layouts per CDNA5 ISA) ----------
// A operand (16x32, MxK), memory row-major M x K (row stride ld halves).
__device__ __forceinline__ v16h load_tileA(const _Float16* __restrict__ p, int ld, int lane) {
  const int half = lane >> 4, m = lane & 15;
  const _Float16* row = p + (size_t)m * ld + 8 * half;
  v8h lo = *(const v8h*)(row);        // K = 8*half + 0..7
  v8h hi = *(const v8h*)(row + 16);   // K = 16 + 8*half + 0..7
  v16h r;
#pragma unroll
  for (int e = 0; e < 8; ++e) { r[e] = lo[e]; r[e + 8] = hi[e]; }
  return r;
}
// B operand (32x16, KxN), memory row-major N x K (i.e. B transposed; row stride ld halves).
__device__ __forceinline__ v16h load_tileB(const _Float16* __restrict__ p, int ld, int lane) {
  const int half = lane >> 4, n = lane & 15;
  const _Float16* row = p + (size_t)n * ld + 8 * half;   // K = 8*half + 0..7, 16 + 8*half + 0..7
  v8h lo = *(const v8h*)(row);
  v8h hi = *(const v8h*)(row + 16);
  v16h r;
#pragma unroll
  for (int e = 0; e < 8; ++e) { r[e] = lo[e]; r[e + 8] = hi[e]; }
  return r;
}

// ---------- f32 -> f16 convert ----------
__global__ void retnet_cvt_f16(const float* __restrict__ src, _Float16* __restrict__ dst, int n) {
  int i = blockIdx.x * blockDim.x + threadIdx.x;
  if (i < n) dst[i] = (_Float16)src[i];
}

// ---------- QKV projection GEMM: out = xh @ W^T ----------
// One wave computes a 32(M) x 64(N) strip with register double-buffering over K.
// mode 0: store (b,h,t,d).  mode 1: store (b,h,d,t) (v transposed, contiguous b128 stores).
__global__ __launch_bounds__(256) void retnet_gemm_qkv(const _Float16* __restrict__ A,
                                                       const _Float16* __restrict__ W,
                                                       _Float16* __restrict__ out, int mode) {
  const int lane = threadIdx.x & 31;
  const int gid  = blockIdx.x * 8 + (threadIdx.x >> 5);   // 2048 waves
  const int m_base = (gid >> 4) * 32;                     // 128 row groups
  const int n_base = (gid & 15) * 64;                     // 16 column groups

  const _Float16* Arow0 = A + (size_t)m_base * CC;
  const _Float16* Arow1 = A + (size_t)(m_base + 16) * CC;

  v8f acc[2][4] = {};
  v16h a0 = load_tileA(Arow0, CC, lane);
  v16h a1 = load_tileA(Arow1, CC, lane);
  v16h b0 = load_tileB(W + (size_t)(n_base +  0) * CC, CC, lane);
  v16h b1 = load_tileB(W + (size_t)(n_base + 16) * CC, CC, lane);
  v16h b2 = load_tileB(W + (size_t)(n_base + 32) * CC, CC, lane);
  v16h b3 = load_tileB(W + (size_t)(n_base + 48) * CC, CC, lane);

  for (int k = 32; k < CC; k += 32) {
    // prefetch next K-step while current WMMAs execute
    v16h na0 = load_tileA(Arow0 + k, CC, lane);
    v16h na1 = load_tileA(Arow1 + k, CC, lane);
    v16h nb0 = load_tileB(W + (size_t)(n_base +  0) * CC + k, CC, lane);
    v16h nb1 = load_tileB(W + (size_t)(n_base + 16) * CC + k, CC, lane);
    v16h nb2 = load_tileB(W + (size_t)(n_base + 32) * CC + k, CC, lane);
    v16h nb3 = load_tileB(W + (size_t)(n_base + 48) * CC + k, CC, lane);
    acc[0][0] = WMMA_F16(a0, b0, acc[0][0]);
    acc[1][0] = WMMA_F16(a1, b0, acc[1][0]);
    acc[0][1] = WMMA_F16(a0, b1, acc[0][1]);
    acc[1][1] = WMMA_F16(a1, b1, acc[1][1]);
    acc[0][2] = WMMA_F16(a0, b2, acc[0][2]);
    acc[1][2] = WMMA_F16(a1, b2, acc[1][2]);
    acc[0][3] = WMMA_F16(a0, b3, acc[0][3]);
    acc[1][3] = WMMA_F16(a1, b3, acc[1][3]);
    a0 = na0; a1 = na1; b0 = nb0; b1 = nb1; b2 = nb2; b3 = nb3;
  }
  acc[0][0] = WMMA_F16(a0, b0, acc[0][0]);
  acc[1][0] = WMMA_F16(a1, b0, acc[1][0]);
  acc[0][1] = WMMA_F16(a0, b1, acc[0][1]);
  acc[1][1] = WMMA_F16(a1, b1, acc[1][1]);
  acc[0][2] = WMMA_F16(a0, b2, acc[0][2]);
  acc[1][2] = WMMA_F16(a1, b2, acc[1][2]);
  acc[0][3] = WMMA_F16(a0, b3, acc[0][3]);
  acc[1][3] = WMMA_F16(a1, b3, acc[1][3]);

  const int half = lane >> 4, n = lane & 15;
#pragma unroll
  for (int ms = 0; ms < 2; ++ms) {
    const int mb = m_base + ms * 16;
    if (mode == 0) {                    // (b,h,t,d)
#pragma unroll
      for (int nt = 0; nt < 4; ++nt) {
        const int o = n_base + nt * 16 + n, h = o >> 6, dim = o & 63;
#pragma unroll
        for (int r = 0; r < 8; ++r) {
          const int tg = mb + r + 8 * half;
          const int b = tg >> 11, t = tg & (TT - 1);
          out[((size_t)(b * HH + h) * TT + t) * DD + dim] = (_Float16)acc[ms][nt][r];
        }
      }
    } else {                            // (b,h,d,t): 8 consecutive t -> one b128 store
#pragma unroll
      for (int nt = 0; nt < 4; ++nt) {
        const int o = n_base + nt * 16 + n, h = o >> 6, dim = o & 63;
        const int tg0 = mb + 8 * half;
        const int b = tg0 >> 11, t0 = tg0 & (TT - 1);
        v8h pk;
#pragma unroll
        for (int r = 0; r < 8; ++r) pk[r] = (_Float16)acc[ms][nt][r];
        *(v8h*)(out + ((size_t)(b * HH + h) * DD + dim) * TT + t0) = pk;
      }
    }
  }
}

// ---------- retention core: y = (Q K^T * D) V, causal decay ----------
// j-loop runs DESCENDING from the diagonal so gamma^(i-j) can be carried as a
// running product (one mul per element per chunk instead of exp2 per element).
__global__ __launch_bounds__(256) void retnet_retention(const _Float16* __restrict__ Q,
                                                        const _Float16* __restrict__ K,
                                                        const _Float16* __restrict__ VT,
                                                        float* __restrict__ Y,
                                                        float* __restrict__ part) {
  __shared__ _Float16 smem[8][16 * 32];   // per-wave 16x32 f16 score staging (1 KB/wave)
  const int lane = threadIdx.x & 31;
  const int wave = threadIdx.x >> 5;
  const int gid  = blockIdx.x * 8 + wave;  // 4096 waves = 32 bh * 128 row tiles
  const int itile = gid & (ITILES - 1);
  const int bh    = gid >> 7;
  const int h     = bh & (HH - 1);
  const int i_base = itile * 16;

  // gamma_h = 1 - exp(linspace(log(1e-3), log(0.1), H))
  const float glo = logf(1.0f - 0.999f), ghi = logf(1.0f - 0.9f);
  const float gamma = 1.0f - expf(glo + (ghi - glo) * ((float)h / (float)(HH - 1)));
  const float l2g = log2f(gamma);
  const float g32 = exp2f(32.0f * l2g);          // gamma^32

  const int half = lane >> 4, n = lane & 15;
  const _Float16* qrow = Q + ((size_t)bh * TT + i_base) * DD;
  const v16h aQ0 = load_tileA(qrow, DD, lane);         // K-dim 0..31
  const v16h aQ1 = load_tileA(qrow + 32, DD, lane);    // K-dim 32..63

  // running decay gamma^(i-j) per element, initialized at the diagonal chunk
  const int j_top = (i_base >> 5) << 5;
  float p0[8], p1[8];
#pragma unroll
  for (int r = 0; r < 8; ++r) {
    const int ig = i_base + r + 8 * half;
    p0[r] = exp2f((float)(ig - (j_top + n))      * l2g);
    p1[r] = exp2f((float)(ig - (j_top + 16 + n)) * l2g);
  }

  v8f acc[4] = {};
  _Float16* L = &smem[wave][0];

  for (int j_base = j_top; j_base >= 0; j_base -= 32) {
    // issue K tiles, then V tiles (LOADcnt retires in order, so the score WMMAs'
    // wait releases while the V loads are still in flight)
    const _Float16* krow = K + ((size_t)bh * TT + j_base) * DD;
    v16h b00 = load_tileB(krow, DD, lane);
    v16h b01 = load_tileB(krow + 32, DD, lane);
    v16h b10 = load_tileB(krow + (size_t)16 * DD, DD, lane);
    v16h b11 = load_tileB(krow + (size_t)16 * DD + 32, DD, lane);
    v16h bV[4];
#pragma unroll
    for (int nt = 0; nt < 4; ++nt) {
      const _Float16* vrow = VT + ((size_t)bh * DD + nt * 16) * TT + j_base;
      bV[nt] = load_tileB(vrow, TT, lane);
    }

    v8f s0 = {}; s0 = WMMA_F16(aQ0, b00, s0); s0 = WMMA_F16(aQ1, b01, s0);
    v8f s1 = {}; s1 = WMMA_F16(aQ0, b10, s1); s1 = WMMA_F16(aQ1, b11, s1);

    // causal decay mask + stage masked scores as f16 row-major in LDS
#pragma unroll
    for (int r = 0; r < 8; ++r) {
      const int ig = i_base + r + 8 * half;
      const int d0 = ig - (j_base + n);
      const int d1 = d0 - 16;
      const float m0 = (d0 < 0) ? 0.0f : p0[r];
      const float m1 = (d1 < 0) ? 0.0f : p1[r];
      p0[r] *= g32;                              // shift decay for next (smaller) j chunk
      p1[r] *= g32;
      L[(r + 8 * half) * 32 + n]      = (_Float16)(s0[r] * m0);
      L[(r + 8 * half) * 32 + 16 + n] = (_Float16)(s1[r] * m1);
    }
    // cross-lane DS dependency within the wave: wait for all stores before reload
    asm volatile("s_wait_dscnt 0" ::: "memory");
    v16h aS = load_tileA(L, 32, lane);

    // y += S_f16 @ V  (V stored transposed (d,t) row-major == NxK for B operand)
#pragma unroll
    for (int nt = 0; nt < 4; ++nt)
      acc[nt] = WMMA_F16(aS, bV[nt], acc[nt]);
  }

  // write raw y (f32) and deterministic per-tile sum / sumsq partials for GroupNorm
  float s = 0.0f, sq = 0.0f;
#pragma unroll
  for (int nt = 0; nt < 4; ++nt) {
    float* yrow = Y + ((size_t)bh * TT + i_base) * DD + nt * 16 + n;
#pragma unroll
    for (int r = 0; r < 8; ++r) {
      const float v = acc[nt][r];
      yrow[(size_t)(r + 8 * half) * DD] = v;
      s += v; sq += v * v;
    }
  }
#pragma unroll
  for (int off = 16; off > 0; off >>= 1) {
    s  += __shfl_xor(s,  off, 32);
    sq += __shfl_xor(sq, off, 32);
  }
  if (lane == 0) {
    part[(size_t)(bh * ITILES + itile) * 2 + 0] = s;
    part[(size_t)(bh * ITILES + itile) * 2 + 1] = sq;
  }
}

// ---------- GroupNorm statistics: deterministic fixed-order reduction ----------
__global__ void retnet_stats(const float* __restrict__ part, float* __restrict__ stat) {
  const int bh = threadIdx.x;
  if (bh >= BB * HH) return;
  float s = 0.0f, sq = 0.0f;
  for (int i = 0; i < ITILES; ++i) {
    s  += part[(size_t)(bh * ITILES + i) * 2 + 0];
    sq += part[(size_t)(bh * ITILES + i) * 2 + 1];
  }
  const float inv = 1.0f / (float)(TT * DD);
  const float mean = s * inv;
  const float var  = sq * inv - mean * mean;
  stat[bh * 2 + 0] = mean;
  stat[bh * 2 + 1] = rsqrtf(var + 1e-5f);
}

// ---------- normalize + affine, repack (b,h,t,d) f32 -> (b*t, c) f16 ----------
__global__ void retnet_norm(const float* __restrict__ Y, const float* __restrict__ stat,
                            const float* __restrict__ gw, const float* __restrict__ gb,
                            _Float16* __restrict__ yn) {
  const int i = blockIdx.x * blockDim.x + threadIdx.x;   // over B*H*T*D = 4M
  if (i >= BB * HH * TT * DD) return;
  const int dim = i & 63;
  const int t   = (i >> 6) & (TT - 1);
  const int h   = (i >> 17) & (HH - 1);
  const int b   = i >> 21;
  const int bh  = b * HH + h;
  const int c   = h * DD + dim;
  const float v = (Y[i] - stat[bh * 2]) * stat[bh * 2 + 1] * gw[c] + gb[c];
  yn[((size_t)(b * TT + t)) * CC + c] = (_Float16)v;
}

// ---------- output projection: out = yn @ wo^T (f32 store), 32x64 per wave ----------
__global__ __launch_bounds__(256) void retnet_gemm_out(const _Float16* __restrict__ A,
                                                       const _Float16* __restrict__ W,
                                                       float* __restrict__ out) {
  const int lane = threadIdx.x & 31;
  const int gid  = blockIdx.x * 8 + (threadIdx.x >> 5);   // 2048 waves
  const int m_base = (gid >> 4) * 32;
  const int n_base = (gid & 15) * 64;

  const _Float16* Arow0 = A + (size_t)m_base * CC;
  const _Float16* Arow1 = A + (size_t)(m_base + 16) * CC;

  v8f acc[2][4] = {};
  v16h a0 = load_tileA(Arow0, CC, lane);
  v16h a1 = load_tileA(Arow1, CC, lane);
  v16h b0 = load_tileB(W + (size_t)(n_base +  0) * CC, CC, lane);
  v16h b1 = load_tileB(W + (size_t)(n_base + 16) * CC, CC, lane);
  v16h b2 = load_tileB(W + (size_t)(n_base + 32) * CC, CC, lane);
  v16h b3 = load_tileB(W + (size_t)(n_base + 48) * CC, CC, lane);

  for (int k = 32; k < CC; k += 32) {
    v16h na0 = load_tileA(Arow0 + k, CC, lane);
    v16h na1 = load_tileA(Arow1 + k, CC, lane);
    v16h nb0 = load_tileB(W + (size_t)(n_base +  0) * CC + k, CC, lane);
    v16h nb1 = load_tileB(W + (size_t)(n_base + 16) * CC + k, CC, lane);
    v16h nb2 = load_tileB(W + (size_t)(n_base + 32) * CC + k, CC, lane);
    v16h nb3 = load_tileB(W + (size_t)(n_base + 48) * CC + k, CC, lane);
    acc[0][0] = WMMA_F16(a0, b0, acc[0][0]);
    acc[1][0] = WMMA_F16(a1, b0, acc[1][0]);
    acc[0][1] = WMMA_F16(a0, b1, acc[0][1]);
    acc[1][1] = WMMA_F16(a1, b1, acc[1][1]);
    acc[0][2] = WMMA_F16(a0, b2, acc[0][2]);
    acc[1][2] = WMMA_F16(a1, b2, acc[1][2]);
    acc[0][3] = WMMA_F16(a0, b3, acc[0][3]);
    acc[1][3] = WMMA_F16(a1, b3, acc[1][3]);
    a0 = na0; a1 = na1; b0 = nb0; b1 = nb1; b2 = nb2; b3 = nb3;
  }
  acc[0][0] = WMMA_F16(a0, b0, acc[0][0]);
  acc[1][0] = WMMA_F16(a1, b0, acc[1][0]);
  acc[0][1] = WMMA_F16(a0, b1, acc[0][1]);
  acc[1][1] = WMMA_F16(a1, b1, acc[1][1]);
  acc[0][2] = WMMA_F16(a0, b2, acc[0][2]);
  acc[1][2] = WMMA_F16(a1, b2, acc[1][2]);
  acc[0][3] = WMMA_F16(a0, b3, acc[0][3]);
  acc[1][3] = WMMA_F16(a1, b3, acc[1][3]);

  const int half = lane >> 4, n = lane & 15;
#pragma unroll
  for (int ms = 0; ms < 2; ++ms) {
#pragma unroll
    for (int nt = 0; nt < 4; ++nt) {
#pragma unroll
      for (int r = 0; r < 8; ++r)
        out[(size_t)(m_base + ms * 16 + r + 8 * half) * CC + n_base + nt * 16 + n] =
            acc[ms][nt][r];
    }
  }
}

extern "C" void kernel_launch(void* const* d_in, const int* in_sizes, int n_in,
                              void* d_out, int out_size, void* d_ws, size_t ws_size,
                              hipStream_t stream) {
  const float* x  = (const float*)d_in[0];
  const float* wq = (const float*)d_in[1];
  const float* wk = (const float*)d_in[2];
  const float* wv = (const float*)d_in[3];
  const float* wo = (const float*)d_in[4];
  const float* gw = (const float*)d_in[5];
  const float* gb = (const float*)d_in[6];

  char* ws = (char*)d_ws;
  _Float16* xh  = (_Float16*)(ws + OFF_XH);
  _Float16* wqh = (_Float16*)(ws + OFF_WQ);
  _Float16* wkh = (_Float16*)(ws + OFF_WK);
  _Float16* wvh = (_Float16*)(ws + OFF_WV);
  _Float16* woh = (_Float16*)(ws + OFF_WO);
  _Float16* qh  = (_Float16*)(ws + OFF_Q);
  _Float16* kh  = (_Float16*)(ws + OFF_K);
  _Float16* vth = (_Float16*)(ws + OFF_VT);
  float*    yr  = (float*)(ws + OFF_Y);
  float*    prt = (float*)(ws + OFF_PART);
  float*    stt = (float*)(ws + OFF_STAT);

  // 1) f32 -> f16 conversions
  retnet_cvt_f16<<<(BT * CC + 255) / 256, 256, 0, stream>>>(x, xh, BT * CC);
  retnet_cvt_f16<<<(CC * CC + 255) / 256, 256, 0, stream>>>(wq, wqh, CC * CC);
  retnet_cvt_f16<<<(CC * CC + 255) / 256, 256, 0, stream>>>(wk, wkh, CC * CC);
  retnet_cvt_f16<<<(CC * CC + 255) / 256, 256, 0, stream>>>(wv, wvh, CC * CC);
  retnet_cvt_f16<<<(CC * CC + 255) / 256, 256, 0, stream>>>(wo, woh, CC * CC);

  // 2) projections (WMMA): q,k head-major; v transposed per head
  retnet_gemm_qkv<<<256, 256, 0, stream>>>(xh, wqh, qh, 0);
  retnet_gemm_qkv<<<256, 256, 0, stream>>>(xh, wkh, kh, 0);
  retnet_gemm_qkv<<<256, 256, 0, stream>>>(xh, wvh, vth, 1);

  // 3) retention core (WMMA, causal decay), emits y + GroupNorm partials
  retnet_retention<<<512, 256, 0, stream>>>(qh, kh, vth, yr, prt);

  // 4) per-(b,h) mean/rstd
  retnet_stats<<<1, 32, 0, stream>>>(prt, stt);

  // 5) GroupNorm + affine; repack to (b*t, c) f16 (reuses xh region)
  _Float16* ynh = xh;
  retnet_norm<<<(BB * HH * TT * DD + 255) / 256, 256, 0, stream>>>(yr, stt, gw, gb, ynh);

  // 6) output projection (WMMA) straight into d_out (f32)
  retnet_gemm_out<<<256, 256, 0, stream>>>(ynh, woh, (float*)d_out);
}
